// Regularization_66812511256711
// MI455X (gfx1250) — matrix operand order repaired and checked by reading, not compiled, not run on hardware
//
#include <hip/hip_runtime.h>

// ---------------------------------------------------------------------------
// Regularization block for MI455X (gfx1250, wave32, WMMA).
// Conv chain: implicit GEMM, v_wmma_f32_16x16x32_bf16, double-buffered LDS
// fed by GLOBAL_LOAD_ASYNC_TO_LDS (ASYNCcnt) when the toolchain exposes it.
// ---------------------------------------------------------------------------

typedef __bf16 bf16_t;
typedef __attribute__((ext_vector_type(16))) __bf16 v16bf;
typedef __attribute__((ext_vector_type(8)))  __bf16 v8bf;
typedef __attribute__((ext_vector_type(4)))  __bf16 v4bf;
typedef __attribute__((ext_vector_type(8)))  float  v8f;
typedef __attribute__((ext_vector_type(4)))  int    v4i;
typedef __attribute__((ext_vector_type(2)))  int    v2i;

#define AS_GLOBAL __attribute__((address_space(1)))
#define AS_LOCAL  __attribute__((address_space(3)))

#if __has_builtin(__builtin_amdgcn_global_load_async_to_lds_b128) && \
    __has_builtin(__builtin_amdgcn_global_load_async_to_lds_b64)
#define HAS_ASYNC_LDS 1
#else
#define HAS_ASYNC_LDS 0
#endif

__device__ __forceinline__ void wait_async0() {
#if __has_builtin(__builtin_amdgcn_s_wait_asynccnt)
    __builtin_amdgcn_s_wait_asynccnt(0);
#else
    asm volatile("s_wait_asynccnt 0" ::: "memory");
#endif
}

constexpr int IMG_B = 4;
constexpr int IMG_H = 192;
constexpr int IMG_W = 448;
constexpr int NPIX  = IMG_B * IMG_H * IMG_W;   // 344064 (multiple of 256 and 64)

// ---------------------------------------------------------------------------
// Implicit-GEMM conv, NHWC bf16 in / NHWC bf16 out, f32 accumulate via WMMA.
//   in   : [NPIX][CIN] bf16 (channel-contiguous)
//   wgt  : [KH*KW][COUT][CIN] bf16 (pre-packed, CIN padded to mult of 32)
//   zeros: >=64B of zero bf16 (redirect target for out-of-bounds taps)
//   out  : [NPIX][outStride] bf16, written at channel offset outOffset
// Tile: 64 pixels (one row segment) x TN output channels per block; 8 waves.
// Double-buffered LDS stages of 32 input channels, async global->LDS copies.
// ---------------------------------------------------------------------------
template <int CIN, int COUT, int TN, int KH, int KW>
__global__ __launch_bounds__(256) void conv_wmma_kernel(
    const bf16_t* __restrict__ in, const bf16_t* __restrict__ wgt,
    const float* __restrict__ bias, const bf16_t* __restrict__ zeros,
    bf16_t* __restrict__ out, int outStride, int outOffset)
{
    constexpr int TM      = 64;
    constexpr int TILES_X = IMG_W / TM;        // 7
    constexpr int NACC    = TN / 32;           // 2 (TN=64) or 1 (TN=32)
    constexpr int NSTEP   = KH * KW * (CIN / 32);
    constexpr int AST     = 40;                // padded LDS row stride (elems)

    const int tid  = threadIdx.x;
    const int lane = tid & 31;
    const int wave = tid >> 5;

    const int t     = blockIdx.x;
    const int x0    = (t % TILES_X) * TM;
    const int row   = t / TILES_X;
    const int y     = row % IMG_H;
    const int b     = row / IMG_H;
    const int cout0 = blockIdx.y * TN;

    __shared__ alignas(16) bf16_t Asmem[2][TM * AST];
    __shared__ alignas(16) bf16_t Bsmem[2][TN * AST];

    v8f acc[NACC];
#pragma unroll
    for (int j = 0; j < NACC; ++j)
#pragma unroll
        for (int e = 0; e < 8; ++e) acc[j][e] = 0.0f;

    const int psub     = wave & 3;             // pixel sub-tile (16 px)
    const int csubBase = (wave >> 2) * NACC;   // cout sub-tile base

    // staging assignments (one async load per thread per tile)
    const int a_px = tid >> 2;
    const int a_co = (tid & 3) * 8;
    const int b_n  = (TN == 64) ? (tid >> 2) : (tid >> 3);
    const int b_co = (TN == 64) ? ((tid & 3) * 8) : ((tid & 7) * 4);

    const int mloc  = lane & 15;   // M for A-frag, N for B/C/D frags
    const int khalf = lane >> 4;

    // per-thread base pointers
    const bf16_t* aBase =
        in + ((long long)(b * IMG_H + y) * IMG_W + x0 + a_px) * CIN + a_co;
    const bf16_t* bBase = wgt + (size_t)(cout0 + b_n) * CIN + b_co;

    // incrementally maintained issue-side indices
    int ky_n = -(KH / 2), kx_n = -(KW / 2), kc_n = 0;
    size_t wTap_n = 0;

    auto issueStage = [&](int buf) {
        const int yy = y + ky_n;
        const int xx = x0 + a_px + kx_n;
        const bool ok = (yy >= 0) && (yy < IMG_H) && (xx >= 0) && (xx < IMG_W);
        const bf16_t* ap =
            ok ? (aBase + ((long long)ky_n * IMG_W + kx_n) * CIN + kc_n) : zeros;
        const bf16_t* bp = bBase + wTap_n + kc_n;
#if HAS_ASYNC_LDS
        __builtin_amdgcn_global_load_async_to_lds_b128(
            (AS_GLOBAL v4i*)ap,
            (AS_LOCAL v4i*)&Asmem[buf][a_px * AST + a_co], 0, 0);
        if constexpr (TN == 64) {
            __builtin_amdgcn_global_load_async_to_lds_b128(
                (AS_GLOBAL v4i*)bp,
                (AS_LOCAL v4i*)&Bsmem[buf][b_n * AST + b_co], 0, 0);
        } else {
            __builtin_amdgcn_global_load_async_to_lds_b64(
                (AS_GLOBAL v2i*)bp,
                (AS_LOCAL v2i*)&Bsmem[buf][b_n * AST + b_co], 0, 0);
        }
#else
        *(v8bf*)&Asmem[buf][a_px * AST + a_co] = *(const v8bf*)ap;
        if constexpr (TN == 64) {
            *(v8bf*)&Bsmem[buf][b_n * AST + b_co] = *(const v8bf*)bp;
        } else {
            *(v4bf*)&Bsmem[buf][b_n * AST + b_co] = *(const v4bf*)bp;
        }
#endif
        // advance (kc fastest, then kx, then ky)
        kc_n += 32;
        if (kc_n == CIN) {
            kc_n = 0;
            kx_n++;
            wTap_n += (size_t)COUT * CIN;
            if (kx_n > KW / 2) { kx_n = -(KW / 2); ky_n++; }
        }
    };

    // compute-side LDS offsets
    const int aRow  = (psub * 16 + mloc) * AST;
    const int aOff0 = aRow + khalf * 8;
    const int aOff1 = aRow + 16 + khalf * 8;

    issueStage(0);
#pragma unroll 1
    for (int st = 0; st < NSTEP; ++st) {
        const int buf = st & 1;
#if HAS_ASYNC_LDS
        wait_async0();          // my stage-st copies have landed in LDS
#endif
        __syncthreads();        // everyone's stage-st data visible; stage st-1 fully consumed
        if (st + 1 < NSTEP) issueStage(buf ^ 1);

        // ---- A fragment (16x32 bf16, documented wave32 layout) ----
        const v8bf a0 = *(const v8bf*)&Asmem[buf][aOff0];
        const v8bf a1 = *(const v8bf*)&Asmem[buf][aOff1];
        const v16bf afrag = __builtin_shufflevector(
            a0, a1, 0, 1, 2, 3, 4, 5, 6, 7, 8, 9, 10, 11, 12, 13, 14, 15);
#pragma unroll
        for (int j = 0; j < NACC; ++j) {
            const int nRow = ((csubBase + j) * 16 + mloc) * AST + khalf * 8;
            const v8bf b0 = *(const v8bf*)&Bsmem[buf][nRow];
            const v8bf b1 = *(const v8bf*)&Bsmem[buf][nRow + 16];
            const v16bf bfrag = __builtin_shufflevector(
                b0, b1, 0, 1, 2, 3, 4, 5, 6, 7, 8, 9, 10, 11, 12, 13, 14, 15);
            acc[j] = __builtin_amdgcn_wmma_f32_16x16x32_bf16(
                false, afrag, false, bfrag, (short)0, acc[j], false, false);
        }
    }

    // ---- epilogue: bias + LeakyReLU(0.1) + bf16 store ----
    const long long pixBase = (long long)(b * IMG_H + y) * IMG_W + x0;
#pragma unroll
    for (int j = 0; j < NACC; ++j) {
        const int n  = cout0 + (csubBase + j) * 16 + mloc;
        const float bv = bias[n];
#pragma unroll
        for (int r = 0; r < 8; ++r) {
            const int m = r + 8 * khalf;
            float v = acc[j][r] + bv;
            v = (v >= 0.0f) ? v : 0.1f * v;
            out[(pixBase + psub * 16 + m) * (long long)outStride + outOffset + n] = (bf16_t)v;
        }
    }
}

// ---------------------------------------------------------------------------
// Weight repack: OIHW f32 -> [tap][O][Ipad] bf16, zero-padded input channels.
// ---------------------------------------------------------------------------
__global__ __launch_bounds__(256) void pack_weights_kernel(
    const float* __restrict__ w, bf16_t* __restrict__ out,
    int O, int I, int KHW, int Ipad)
{
    int idx = blockIdx.x * 256 + threadIdx.x;
    int total = KHW * O * Ipad;
    if (idx >= total) return;
    int i   = idx % Ipad;
    int rem = idx / Ipad;
    int o   = rem % O;
    int t   = rem / O;
    bf16_t v = (bf16_t)0.0f;
    if (i < I) v = (bf16_t)w[((size_t)o * I + i) * KHW + t];
    out[idx] = v;
}

// ---------------------------------------------------------------------------
// Zero-fill small scratch (OOB redirect target) -- runs every launch.
// ---------------------------------------------------------------------------
__global__ void zero_kernel(float* __restrict__ p)
{
    p[threadIdx.x] = 0.0f;   // 64 threads -> 256 B
}

// ---------------------------------------------------------------------------
// Per-(b, channel) mean of flow over H*W.
// ---------------------------------------------------------------------------
__global__ __launch_bounds__(256) void mean_kernel(
    const float* __restrict__ flow, float* __restrict__ means)
{
    const size_t hw = (size_t)IMG_H * IMG_W;
    const int bc = blockIdx.x;   // 0..7 = b*2 + c
    const float* src = flow + (size_t)bc * hw;
    float s = 0.0f;
    for (size_t i = threadIdx.x; i < hw; i += 256) s += src[i];
    __shared__ float red[256];
    red[threadIdx.x] = s;
    __syncthreads();
    for (int off = 128; off > 0; off >>= 1) {
        if (threadIdx.x < off) red[threadIdx.x] += red[threadIdx.x + off];
        __syncthreads();
    }
    if (threadIdx.x == 0) means[bc] = red[0] / (float)hw;
}

// ---------------------------------------------------------------------------
// Prep: backwarp diff, centered flow, zero-pad tail of X0; transpose feat
// input to NHWC bf16.
// ---------------------------------------------------------------------------
__global__ __launch_bounds__(256) void prep_kernel(
    const float* __restrict__ t1, const float* __restrict__ t2,
    const float* __restrict__ feat1, const float* __restrict__ flow,
    const float* __restrict__ means,
    bf16_t* __restrict__ x0, bf16_t* __restrict__ f1out)
{
    const int p = blockIdx.x * 256 + threadIdx.x;
    const int x  = p % IMG_W;
    const int rr = p / IMG_W;
    const int y  = rr % IMG_H;
    const int b  = rr / IMG_H;
    const size_t hw = (size_t)IMG_H * IMG_W;

    const float* fb = flow + (size_t)b * 2 * hw;
    const float fx = fb[(size_t)y * IMG_W + x];
    const float fy = fb[hw + (size_t)y * IMG_W + x];

    // bilinear backwarp of tensor2 at (x + 5*fx, y + 5*fy), zero padded
    const float gx = (float)x + 5.0f * fx;
    const float gy = (float)y + 5.0f * fy;
    const float xf = floorf(gx), yf = floorf(gy);
    const int ix = (int)xf, iy = (int)yf;
    const float wx1 = gx - xf, wy1 = gy - yf;
    const float wx0 = 1.0f - wx1, wy0 = 1.0f - wy1;

    const float* t2b = t2 + (size_t)b * 3 * hw;
    const float* t1b = t1 + (size_t)b * 3 * hw;
    float diff2 = 0.0f;
#pragma unroll
    for (int c = 0; c < 3; ++c) {
        const float* img = t2b + (size_t)c * hw;
        float v00 = 0.f, v01 = 0.f, v10 = 0.f, v11 = 0.f;
        const bool y0ok = (iy >= 0) && (iy < IMG_H);
        const bool y1ok = (iy + 1 >= 0) && (iy + 1 < IMG_H);
        const bool x0ok = (ix >= 0) && (ix < IMG_W);
        const bool x1ok = (ix + 1 >= 0) && (ix + 1 < IMG_W);
        if (y0ok && x0ok) v00 = img[(size_t)iy * IMG_W + ix];
        if (y0ok && x1ok) v01 = img[(size_t)iy * IMG_W + ix + 1];
        if (y1ok && x0ok) v10 = img[(size_t)(iy + 1) * IMG_W + ix];
        if (y1ok && x1ok) v11 = img[(size_t)(iy + 1) * IMG_W + ix + 1];
        const float w = v00 * wy0 * wx0 + v01 * wy0 * wx1 +
                        v10 * wy1 * wx0 + v11 * wy1 * wx1;
        const float d = t1b[(size_t)c * hw + (size_t)y * IMG_W + x] - w;
        diff2 += d * d;
    }

    bf16_t* xo = x0 + (size_t)p * 160;
    xo[0] = (bf16_t)sqrtf(diff2);
    xo[1] = (bf16_t)(fx - means[b * 2 + 0]);
    xo[2] = (bf16_t)(fy - means[b * 2 + 1]);
#pragma unroll
    for (int c = 131; c < 160; ++c) xo[c] = (bf16_t)0.0f;

    const float* ftb = feat1 + (size_t)b * 64 * hw + (size_t)y * IMG_W + x;
    bf16_t* fo = f1out + (size_t)p * 64;
#pragma unroll 8
    for (int c = 0; c < 64; ++c) fo[c] = (bf16_t)ftb[(size_t)c * hw];
}

// ---------------------------------------------------------------------------
// dist1: 5x1 conv (pad 2 vertical), 32 -> 25 channels, f32 out [NPIX][25].
// ---------------------------------------------------------------------------
__global__ __launch_bounds__(256) void dist1_kernel(
    const bf16_t* __restrict__ x6, const float* __restrict__ w1,
    const float* __restrict__ b1, float* __restrict__ out)
{
    __shared__ float ws[25 * 32 * 5];
    for (int i = threadIdx.x; i < 25 * 32 * 5; i += 256) ws[i] = w1[i];
    __syncthreads();

    const int p = blockIdx.x * 256 + threadIdx.x;
    const int x  = p % IMG_W;
    const int rr = p / IMG_W;
    const int y  = rr % IMG_H;
    const int b  = rr / IMG_H;

    float acc[25];
#pragma unroll
    for (int c = 0; c < 25; ++c) acc[c] = b1[c];

#pragma unroll 1
    for (int tt = 0; tt < 5; ++tt) {
        const int yy = y + tt - 2;
        if (yy < 0 || yy >= IMG_H) continue;
        const bf16_t* src = x6 + ((size_t)(b * IMG_H + yy) * IMG_W + x) * 32;
        float xv[32];
#pragma unroll
        for (int i = 0; i < 32; ++i) xv[i] = (float)src[i];
#pragma unroll
        for (int c = 0; c < 25; ++c) {
            float s = acc[c];
#pragma unroll
            for (int i = 0; i < 32; ++i) s += ws[(c * 32 + i) * 5 + tt] * xv[i];
            acc[c] = s;
        }
    }
    float* o = out + (size_t)p * 25;
#pragma unroll
    for (int c = 0; c < 25; ++c) o[c] = acc[c];
}

// ---------------------------------------------------------------------------
// dist2 (1x5 conv, 25->25) + -(d^2) + softmax + unfold-weighted flow sums
// + sx/sy 1x1 convs + divisor. Writes final output [B,2,H,W] f32.
// ---------------------------------------------------------------------------
__global__ __launch_bounds__(256) void dist2_final_kernel(
    const float* __restrict__ d1, const float* __restrict__ w2,
    const float* __restrict__ b2, const float* __restrict__ flow,
    const float* __restrict__ sxw, const float* __restrict__ sxb,
    const float* __restrict__ syw, const float* __restrict__ syb,
    float* __restrict__ out)
{
    __shared__ float ws[25 * 25 * 5];
    for (int i = threadIdx.x; i < 25 * 25 * 5; i += 256) ws[i] = w2[i];
    __shared__ float swx[25], swy[25];
    if (threadIdx.x < 25) { swx[threadIdx.x] = sxw[threadIdx.x]; swy[threadIdx.x] = syw[threadIdx.x]; }
    __syncthreads();

    const int p = blockIdx.x * 256 + threadIdx.x;
    const int x  = p % IMG_W;
    const int rr = p / IMG_W;
    const int y  = rr % IMG_H;
    const int b  = rr / IMG_H;

    float acc[25];
#pragma unroll
    for (int c = 0; c < 25; ++c) acc[c] = b2[c];

#pragma unroll 1
    for (int tt = 0; tt < 5; ++tt) {
        const int xx = x + tt - 2;
        if (xx < 0 || xx >= IMG_W) continue;
        const float* v = d1 + (size_t)(p + tt - 2) * 25;   // same image row
        float vv[25];
#pragma unroll
        for (int j = 0; j < 25; ++j) vv[j] = v[j];
#pragma unroll
        for (int c = 0; c < 25; ++c) {
            float s = acc[c];
#pragma unroll
            for (int j = 0; j < 25; ++j) s += ws[(c * 25 + j) * 5 + tt] * vv[j];
            acc[c] = s;
        }
    }

    float mx = -3.0e38f;
#pragma unroll
    for (int c = 0; c < 25; ++c) { acc[c] = -(acc[c] * acc[c]); mx = fmaxf(mx, acc[c]); }
    float ssum = 0.0f;
#pragma unroll
    for (int c = 0; c < 25; ++c) { acc[c] = __expf(acc[c] - mx); ssum += acc[c]; }
    const float inv = 1.0f / ssum;

    const size_t hw = (size_t)IMG_H * IMG_W;
    const float* f0 = flow + (size_t)b * 2 * hw;
    float sx = sxb[0], sy = syb[0];
#pragma unroll
    for (int k = 0; k < 25; ++k) {
        const int yy = y + k / 5 - 2;
        const int xx = x + (k % 5) - 2;
        float u1 = 0.0f, u2 = 0.0f;
        if (yy >= 0 && yy < IMG_H && xx >= 0 && xx < IMG_W) {
            u1 = f0[(size_t)yy * IMG_W + xx];
            u2 = f0[hw + (size_t)yy * IMG_W + xx];
        }
        sx += swx[k] * acc[k] * u1;
        sy += swy[k] * acc[k] * u2;
    }
    const size_t ob = (size_t)b * 2 * hw + (size_t)y * IMG_W + x;
    out[ob]      = sx * inv;
    out[ob + hw] = sy * inv;
}

// ---------------------------------------------------------------------------
// Host side
// ---------------------------------------------------------------------------
extern "C" void kernel_launch(void* const* d_in, const int* in_sizes, int n_in,
                              void* d_out, int out_size, void* d_ws, size_t ws_size,
                              hipStream_t stream)
{
    const float* tensor1 = (const float*)d_in[0];
    const float* tensor2 = (const float*)d_in[1];
    const float* feat1   = (const float*)d_in[2];
    // d_in[3] (features_tensor2) unused by reference
    const float* flow    = (const float*)d_in[4];
    const float* feat_w  = (const float*)d_in[5];
    const float* feat_b  = (const float*)d_in[6];
    const float* main_w1 = (const float*)d_in[7];
    const float* main_b1 = (const float*)d_in[8];
    const float* main_w2 = (const float*)d_in[9];
    const float* main_b2 = (const float*)d_in[10];
    const float* main_w3 = (const float*)d_in[11];
    const float* main_b3 = (const float*)d_in[12];
    const float* main_w4 = (const float*)d_in[13];
    const float* main_b4 = (const float*)d_in[14];
    const float* main_w5 = (const float*)d_in[15];
    const float* main_b5 = (const float*)d_in[16];
    const float* main_w6 = (const float*)d_in[17];
    const float* main_b6 = (const float*)d_in[18];
    const float* dist_w1 = (const float*)d_in[19];
    const float* dist_b1 = (const float*)d_in[20];
    const float* dist_w2 = (const float*)d_in[21];
    const float* dist_b2 = (const float*)d_in[22];
    const float* sx_w    = (const float*)d_in[23];
    const float* sx_b    = (const float*)d_in[24];
    const float* sy_w    = (const float*)d_in[25];
    const float* sy_b    = (const float*)d_in[26];

    // workspace carve-up
    char* ws = (char*)d_ws;
    size_t off = 0;
    auto carve = [&](size_t bytes) -> void* {
        void* p = ws + off;
        off += (bytes + 255) & ~(size_t)255;
        return p;
    };
    bf16_t* buf0  = (bf16_t*)carve((size_t)NPIX * 160 * sizeof(bf16_t)); // X0/X2/X4/X6
    bf16_t* buf1  = (bf16_t*)carve((size_t)NPIX * 128 * sizeof(bf16_t)); // F1/X1/X3/X5/dist1
    bf16_t* Wfeat = (bf16_t*)carve((size_t)1 * 128 * 64 * sizeof(bf16_t));
    bf16_t* W1    = (bf16_t*)carve((size_t)9 * 128 * 160 * sizeof(bf16_t));
    bf16_t* W2    = (bf16_t*)carve((size_t)9 * 128 * 128 * sizeof(bf16_t));
    bf16_t* W3    = (bf16_t*)carve((size_t)9 * 64 * 128 * sizeof(bf16_t));
    bf16_t* W4    = (bf16_t*)carve((size_t)9 * 64 * 64 * sizeof(bf16_t));
    bf16_t* W5    = (bf16_t*)carve((size_t)9 * 32 * 64 * sizeof(bf16_t));
    bf16_t* W6    = (bf16_t*)carve((size_t)9 * 32 * 32 * sizeof(bf16_t));
    float*  means = (float*)carve(8 * sizeof(float));
    bf16_t* zbuf  = (bf16_t*)carve(256);
    (void)ws_size; (void)in_sizes; (void)n_in; (void)out_size;

    auto pack = [&](const float* w, bf16_t* o, int O, int I, int KHW, int Ipad) {
        int total = KHW * O * Ipad;
        pack_weights_kernel<<<(total + 255) / 256, 256, 0, stream>>>(w, o, O, I, KHW, Ipad);
    };
    pack(feat_w,  Wfeat, 128, 64,  1, 64);
    pack(main_w1, W1,    128, 131, 9, 160);
    pack(main_w2, W2,    128, 128, 9, 128);
    pack(main_w3, W3,    64,  128, 9, 128);
    pack(main_w4, W4,    64,  64,  9, 64);
    pack(main_w5, W5,    32,  64,  9, 64);
    pack(main_w6, W6,    32,  32,  9, 32);

    zero_kernel<<<1, 64, 0, stream>>>((float*)zbuf);
    mean_kernel<<<8, 256, 0, stream>>>(flow, means);
    prep_kernel<<<NPIX / 256, 256, 0, stream>>>(tensor1, tensor2, feat1, flow,
                                                means, buf0, buf1);

    const int PT = NPIX / 64;   // 5376 pixel tiles
    // feat 1x1 conv: F1(buf1,64) -> X0(buf0, offset 3, stride 160), lrelu
    conv_wmma_kernel<64, 128, 64, 1, 1><<<dim3(PT, 2), 256, 0, stream>>>(
        buf1, Wfeat, feat_b, zbuf, buf0, 160, 3);
    // main conv chain (all lrelu)
    conv_wmma_kernel<160, 128, 64, 3, 3><<<dim3(PT, 2), 256, 0, stream>>>(
        buf0, W1, main_b1, zbuf, buf1, 128, 0);
    conv_wmma_kernel<128, 128, 64, 3, 3><<<dim3(PT, 2), 256, 0, stream>>>(
        buf1, W2, main_b2, zbuf, buf0, 128, 0);
    conv_wmma_kernel<128, 64, 64, 3, 3><<<dim3(PT, 1), 256, 0, stream>>>(
        buf0, W3, main_b3, zbuf, buf1, 64, 0);
    conv_wmma_kernel<64, 64, 64, 3, 3><<<dim3(PT, 1), 256, 0, stream>>>(
        buf1, W4, main_b4, zbuf, buf0, 64, 0);
    conv_wmma_kernel<64, 32, 32, 3, 3><<<dim3(PT, 1), 256, 0, stream>>>(
        buf0, W5, main_b5, zbuf, buf1, 32, 0);
    conv_wmma_kernel<32, 32, 32, 3, 3><<<dim3(PT, 1), 256, 0, stream>>>(
        buf1, W6, main_b6, zbuf, buf0, 32, 0);

    float* d1buf = (float*)buf1;   // [NPIX][25] f32 (fits in buf1)
    dist1_kernel<<<NPIX / 256, 256, 0, stream>>>(buf0, dist_w1, dist_b1, d1buf);
    dist2_final_kernel<<<NPIX / 256, 256, 0, stream>>>(
        d1buf, dist_w2, dist_b2, flow, sx_w, sx_b, sy_w, sy_b, (float*)d_out);
}
